// Experts_1889785610412
// MI455X (gfx1250) — compile-verified, matrix operand order checked by reading, not run on hardware
//
#include <hip/hip_runtime.h>

// ---------------------------------------------------------------------------
// MoE SwiGLU FFN for MI455X (gfx1250), bf16 WMMA path, double-buffered LDS.
//   kernel 1: mid = silu(x@w1) * (x@w3)   (fp32 in, bf16 out to workspace)
//   kernel 2: out = mid@w2                (bf16 in, fp32 out)
// ---------------------------------------------------------------------------

typedef __attribute__((ext_vector_type(16))) __bf16       v16bf;
typedef __attribute__((ext_vector_type(4)))  __bf16       v4bf;
typedef __attribute__((ext_vector_type(2)))  __bf16       v2bf;
typedef __attribute__((ext_vector_type(8)))  float        v8f;
typedef __attribute__((ext_vector_type(4)))  float        v4f;
typedef __attribute__((ext_vector_type(2)))  float        v2f;
typedef __attribute__((ext_vector_type(4)))  unsigned int v4u;
typedef __attribute__((ext_vector_type(2)))  unsigned int v2u;

#define E_ 8
#define G_ 2048
#define D_ 2048
#define F_ 5632

#define BM 128          // block tile M
#define BN 64           // block tile N
#define BK 32           // K step == WMMA K
#define LDSS 40         // LDS row stride in bf16 elems (BK + 8 pad, 80B: 16B aligned)

struct B128x2 { v4u lo; v4u hi; };   // 32 bytes == one v16bf fragment

// fp32 -> bf16 via native conversions (backend: v_cvt_pk_bf16_f32)
__device__ __forceinline__ unsigned int cvt2(float a, float b) {
    v2f t; t.x = a; t.y = b;
    v2bf r = __builtin_convertvector(t, v2bf);
    return __builtin_bit_cast(unsigned int, r);
}
__device__ __forceinline__ v2u cvt4(v4f v) {
    v4bf r = __builtin_convertvector(v, v4bf);
    return __builtin_bit_cast(v2u, r);
}
__device__ __forceinline__ unsigned short bfbits(float f) {
    __bf16 h = (__bf16)f;
    return __builtin_bit_cast(unsigned short, h);
}

// Load one 16x32 bf16 WMMA fragment from an LDS tile whose rows (M for A,
// N for transposed-B) are LDSS bf16 elements apart. Matches the ISA layout:
//   lane<16 : K 0..7 in v0-3,  K16..23 in v4-7
//   lane>=16: K 8..15 in v0-3, K24..31 in v4-7
__device__ __forceinline__ v16bf load_frag(const unsigned short* lds, int rowBase) {
    const int lane = threadIdx.x & 31;
    const int row  = rowBase + (lane & 15);
    const int kh   = (lane >> 4) << 3;                  // 0 or 8 elements
    const unsigned short* p = lds + row * LDSS + kh;
    B128x2 r;
    r.lo = *reinterpret_cast<const v4u*>(p);            // K =  kh .. kh+7
    r.hi = *reinterpret_cast<const v4u*>(p + 16);       // K = 16+kh .. 16+kh+7
    return __builtin_bit_cast(v16bf, r);
}

// ---------------------------------------------------------------------------
// Kernel 1: gate/up GEMMs + SwiGLU epilogue.  grid = (F/BN, G/BM, E)
// ---------------------------------------------------------------------------
__global__ __launch_bounds__(256)
void gateup_kernel(const float* __restrict__ x,      // [E*G, D]
                   const float* __restrict__ w1,     // [E, D, F]
                   const float* __restrict__ w3,     // [E, D, F]
                   unsigned short* __restrict__ mid) // [E*G, F] bf16 bits
{
    __shared__ unsigned short Alds [2][BM * LDSS];
    __shared__ unsigned short B1lds[2][BN * LDSS];
    __shared__ unsigned short B3lds[2][BN * LDSS];

    const int e     = blockIdx.z;
    const int gBase = e * G_ + blockIdx.y * BM;     // row into [E*G, *]
    const int nBase = blockIdx.x * BN;

    const float* w1e = w1 + (size_t)e * D_ * F_;
    const float* w3e = w3 + (size_t)e * D_ * F_;

    const int tid  = threadIdx.x;
    const int lane = tid & 31;
    const int wave = tid >> 5;
    const int wm   = (wave & 3) * 32;               // wave M offset (4 waves)
    const int wn   = (wave >> 2) * 32;              // wave N offset (2 waves)

    v8f accg[2][2] = {};
    v8f accu[2][2] = {};

    v4f ra[4], rb1[2], rb3[2];                      // global-load staging regs

    const int rA  = tid >> 3, cA = (tid & 7) * 4;   // A: 32 rows/pass, 4 passes
    const int rB2 = (tid >> 4) * 2;                 // B: adjacent K-row pair
    const int cB  = (tid & 15) * 4;                 // B: 4 columns per thread

    auto load_tiles = [&](int k0) {
        #pragma unroll
        for (int p = 0; p < 4; ++p)
            ra[p] = *reinterpret_cast<const v4f*>(
                x + (size_t)(gBase + rA + p * 32) * D_ + k0 + cA);
        #pragma unroll
        for (int p = 0; p < 2; ++p) {               // K rows rB2, rB2+1
            rb1[p] = *reinterpret_cast<const v4f*>(
                w1e + (size_t)(k0 + rB2 + p) * F_ + nBase + cB);
            rb3[p] = *reinterpret_cast<const v4f*>(
                w3e + (size_t)(k0 + rB2 + p) * F_ + nBase + cB);
        }
    };
    auto store_tiles = [&](int buf) {
        #pragma unroll
        for (int p = 0; p < 4; ++p)
            *reinterpret_cast<v2u*>(&Alds[buf][(rA + p * 32) * LDSS + cA]) = cvt4(ra[p]);
        #pragma unroll
        for (int j = 0; j < 4; ++j) {               // transpose to [n][k]: one b32 per col
            *reinterpret_cast<unsigned int*>(&B1lds[buf][(cB + j) * LDSS + rB2]) =
                cvt2(rb1[0][j], rb1[1][j]);
            *reinterpret_cast<unsigned int*>(&B3lds[buf][(cB + j) * LDSS + rB2]) =
                cvt2(rb3[0][j], rb3[1][j]);
        }
    };
    auto compute = [&](int buf) {
        v16bf af[2], b1f[2], b3f[2];
        #pragma unroll
        for (int i = 0; i < 2; ++i) af[i] = load_frag(Alds[buf], wm + i * 16);
        #pragma unroll
        for (int j = 0; j < 2; ++j) {
            b1f[j] = load_frag(B1lds[buf], wn + j * 16);
            b3f[j] = load_frag(B3lds[buf], wn + j * 16);
        }
        #pragma unroll
        for (int i = 0; i < 2; ++i)
            #pragma unroll
            for (int j = 0; j < 2; ++j) {
                accg[i][j] = __builtin_amdgcn_wmma_f32_16x16x32_bf16(
                    false, af[i], false, b1f[j], (short)0, accg[i][j], false, false);
                accu[i][j] = __builtin_amdgcn_wmma_f32_16x16x32_bf16(
                    false, af[i], false, b3f[j], (short)0, accu[i][j], false, false);
            }
    };

    // double-buffered main loop: one barrier per K-step, VMEM overlapped with WMMA
    load_tiles(0);
    store_tiles(0);
    const int steps = D_ / BK;
    for (int s = 1; s < steps; ++s) {
        __syncthreads();
        load_tiles(s * BK);          // issue global loads for step s
        compute((s - 1) & 1);        // WMMA on step s-1 while loads in flight
        store_tiles(s & 1);
    }
    __syncthreads();
    compute((steps - 1) & 1);

    // ---- SwiGLU epilogue, store bf16 mid ----
    const int col0 = nBase + wn + (lane & 15);
    const int rsel = (lane >> 4) * 8;               // C/D layout: lanes 16-31 -> M+8
    #pragma unroll
    for (int i = 0; i < 2; ++i)
        #pragma unroll
        for (int j = 0; j < 2; ++j)
            #pragma unroll
            for (int r = 0; r < 8; ++r) {
                const float g = accg[i][j][r];
                const float u = accu[i][j][r];
                // silu(g) = g * sigmoid(g); fast rcp instead of IEEE divide
                const float s = g * __builtin_amdgcn_rcpf(1.0f + __expf(-g));
                const int row = gBase + wm + i * 16 + rsel + r;
                const int col = col0 + j * 16;
                mid[(size_t)row * F_ + col] = bfbits(s * u);
            }
}

// ---------------------------------------------------------------------------
// Kernel 2: out = mid @ w2.  grid = (D/BN, G/BM, E)
// ---------------------------------------------------------------------------
__global__ __launch_bounds__(256)
void down_kernel(const unsigned short* __restrict__ mid, // [E*G, F] bf16 bits
                 const float* __restrict__ w2,           // [E, F, D]
                 float* __restrict__ out)                 // [E*G, D]
{
    __shared__ unsigned short Alds[2][BM * LDSS];
    __shared__ unsigned short Blds[2][BN * LDSS];

    const int e     = blockIdx.z;
    const int gBase = e * G_ + blockIdx.y * BM;
    const int nBase = blockIdx.x * BN;

    const float* w2e = w2 + (size_t)e * F_ * D_;

    const int tid  = threadIdx.x;
    const int lane = tid & 31;
    const int wave = tid >> 5;
    const int wm   = (wave & 3) * 32;
    const int wn   = (wave >> 2) * 32;

    v8f acc[2][2] = {};

    v4u ra[2];                                      // A is already bf16
    v4f rb[2];

    const int rA  = tid >> 2, cA = (tid & 3) * 8;   // A: 64 rows/pass, 2 passes
    const int rB2 = (tid >> 4) * 2;                 // B: adjacent K-row pair
    const int cB  = (tid & 15) * 4;

    auto load_tiles = [&](int k0) {
        #pragma unroll
        for (int p = 0; p < 2; ++p)
            ra[p] = *reinterpret_cast<const v4u*>(
                mid + (size_t)(gBase + rA + p * 64) * F_ + k0 + cA);
        #pragma unroll
        for (int p = 0; p < 2; ++p)
            rb[p] = *reinterpret_cast<const v4f*>(
                w2e + (size_t)(k0 + rB2 + p) * D_ + nBase + cB);
    };
    auto store_tiles = [&](int buf) {
        #pragma unroll
        for (int p = 0; p < 2; ++p)
            *reinterpret_cast<v4u*>(&Alds[buf][(rA + p * 64) * LDSS + cA]) = ra[p];
        #pragma unroll
        for (int j = 0; j < 4; ++j)                 // transpose to [n][k]
            *reinterpret_cast<unsigned int*>(&Blds[buf][(cB + j) * LDSS + rB2]) =
                cvt2(rb[0][j], rb[1][j]);
    };
    auto compute = [&](int buf) {
        v16bf af[2], bf[2];
        #pragma unroll
        for (int i = 0; i < 2; ++i) af[i] = load_frag(Alds[buf], wm + i * 16);
        #pragma unroll
        for (int j = 0; j < 2; ++j) bf[j] = load_frag(Blds[buf], wn + j * 16);
        #pragma unroll
        for (int i = 0; i < 2; ++i)
            #pragma unroll
            for (int j = 0; j < 2; ++j)
                acc[i][j] = __builtin_amdgcn_wmma_f32_16x16x32_bf16(
                    false, af[i], false, bf[j], (short)0, acc[i][j], false, false);
    };

    load_tiles(0);
    store_tiles(0);
    const int steps = F_ / BK;
    for (int s = 1; s < steps; ++s) {
        __syncthreads();
        load_tiles(s * BK);
        compute((s - 1) & 1);
        store_tiles(s & 1);
    }
    __syncthreads();
    compute((steps - 1) & 1);

    const int col0 = nBase + wn + (lane & 15);
    const int rsel = (lane >> 4) * 8;
    #pragma unroll
    for (int i = 0; i < 2; ++i)
        #pragma unroll
        for (int j = 0; j < 2; ++j)
            #pragma unroll
            for (int r = 0; r < 8; ++r) {
                const int row = gBase + wm + i * 16 + rsel + r;
                const int col = col0 + j * 16;
                out[(size_t)row * D_ + col] = acc[i][j][r];
            }
}

// ---------------------------------------------------------------------------
extern "C" void kernel_launch(void* const* d_in, const int* in_sizes, int n_in,
                              void* d_out, int out_size, void* d_ws, size_t ws_size,
                              hipStream_t stream) {
    (void)in_sizes; (void)n_in; (void)out_size; (void)ws_size;
    const float* x  = (const float*)d_in[0];   // routed_in_egD [E*G, D]
    const float* w1 = (const float*)d_in[1];   // [E, D, F]
    const float* w2 = (const float*)d_in[2];   // [E, F, D]
    const float* w3 = (const float*)d_in[3];   // [E, D, F]
    unsigned short* mid = (unsigned short*)d_ws;   // [E*G, F] bf16 (185 MB)

    dim3 blk(256, 1, 1);
    dim3 g1(F_ / BN, G_ / BM, E_);   // (88, 16, 8)
    dim3 g2(D_ / BN, G_ / BM, E_);   // (32, 16, 8)
    gateup_kernel<<<g1, blk, 0, stream>>>(x, w1, w3, mid);
    down_kernel  <<<g2, blk, 0, stream>>>(mid, w2, (float*)d_out);
}
